// Snake_WTConv2d_10118942949521
// MI455X (gfx1250) — matrix operand order repaired and checked by reading, not compiled, not run on hardware
//
#include <hip/hip_runtime.h>

typedef __attribute__((ext_vector_type(16))) _Float16 v16h;
typedef __attribute__((ext_vector_type(8)))  float    v8f;

#define BB 4
#define CC 64
#define C4V 256

// ---------------------------------------------------------------- helpers
static __device__ __forceinline__ float reflect_clip(float coord, int size) {
  float span = (float)size;
  float a = coord + 0.5f;
  float m = 2.0f * span;
  float r = fmodf(a, m);
  r = (r < 0.0f) ? (r + m) : r;
  r = (r > span) ? (m - r) : r;
  r -= 0.5f;
  return fminf(fmaxf(r, 0.0f), span - 1.0f);
}

// ---------------------------------------------------------------- Haar DWT
// src: (B, 64, Hs, Ws) -> t: (B, 256, Hs/2, Ws/2) band-interleaved (ch = ci*4+band)
// ll (optional): (B, 64, Hs/2, Ws/2) = band0
__global__ __launch_bounds__(256) void wt_kernel(const float* __restrict__ src,
                                                 float* __restrict__ t,
                                                 float* __restrict__ ll,
                                                 int Hs, int Ws) {
  int h2 = Hs >> 1, w2 = Ws >> 1;
  int n = BB * CC * h2 * w2;
  int idx = blockIdx.x * blockDim.x + threadIdx.x;
  if (idx >= n) return;
  int x = idx % w2; int tmp = idx / w2;
  int y = tmp % h2; tmp /= h2;
  int ci = tmp % CC; int bi = tmp / CC;
  const float* sp = src + (((size_t)(bi * CC + ci) * Hs + 2 * y) * Ws + 2 * x);
  float p00 = sp[0], p01 = sp[1], p10 = sp[Ws], p11 = sp[Ws + 1];
  float b0 = 0.5f * (p00 + p01 + p10 + p11);
  float b1 = 0.5f * (p00 + p01 - p10 - p11);
  float b2 = 0.5f * (p00 - p01 + p10 - p11);
  float b3 = 0.5f * (p00 - p01 - p10 + p11);
  size_t plane = (size_t)h2 * w2;
  size_t base = ((size_t)bi * C4V + (size_t)ci * 4) * plane + (size_t)y * w2 + x;
  t[base] = b0; t[base + plane] = b1; t[base + 2 * plane] = b2; t[base + 3 * plane] = b3;
  if (ll) ll[(size_t)(bi * CC + ci) * plane + (size_t)y * w2 + x] = b0;
}

// ---------------------------------------------------------------- offset conv (WMMA)
// t: (B, 256, h2, w2), W: (2, 256, 3, 3), bias: (2,) -> off: (B, 2, h2, w2) = tanh(conv+b)
// GEMM: M = 16 pixels/wave, N = 16 (2 used), K = 2304 in 9 taps x 8 chunks of 32.
// ci0 loop unrolled x4 -> 4 static v_wmma per tap iteration.
__global__ __launch_bounds__(128) void offset_conv_wmma(const float* __restrict__ t,
                                                        const float* __restrict__ W,
                                                        const float* __restrict__ bias,
                                                        float* __restrict__ off,
                                                        int h2, int w2) {
  __shared__ _Float16 wlds[2 * 256 * 9];  // 9 KB of f16 weights
  for (int i = threadIdx.x; i < 2 * 256 * 9; i += blockDim.x)
    wlds[i] = (_Float16)W[i];
  __syncthreads();

  const int lane = threadIdx.x & 31;
  const int wave = threadIdx.x >> 5;
  const int tile = blockIdx.x * 4 + wave;      // one 16-pixel tile per wave (grid exact)
  const int plane = h2 * w2;
  int pix0 = tile * 16;
  int bi = pix0 / plane; int rem = pix0 - bi * plane;
  int y = rem / w2; int x0 = rem - y * w2;     // w2 % 16 == 0 -> row-aligned tile
  const int m = lane & 15;                     // pixel (M) / column (N) sub-index
  const int x = x0 + m;
  const bool hiK = lane >= 16;
  const float* tb = t + (size_t)bi * C4V * plane;

  v8f acc = {};
  for (int tap = 0; tap < 9; ++tap) {
    int dy = tap / 3 - 1, dx = tap % 3 - 1;
    int yy = y + dy, xx = x + dx;
    bool inb = (yy >= 0) && (yy < h2) && (xx >= 0) && (xx < w2);
    const float* tp = tb + (size_t)yy * w2 + xx;
#pragma unroll 4
    for (int ci0 = 0; ci0 < 256; ci0 += 32) {
      // A fragment: 16-bit 16x32 layout (lane<16: K in {0..7,16..23}; lane>=16: +8)
      v16h a;
#pragma unroll
      for (int j = 0; j < 16; ++j) {
        int k = (j < 8) ? j : (j + 8);
        if (hiK) k += 8;
        float v = inb ? tp[(size_t)(ci0 + k) * plane] : 0.0f;
        a[j] = (_Float16)v;
      }
      // B fragment: 32x16; col N = lane%16, rows K from LDS (only N<2 nonzero)
      v16h b;
#pragma unroll
      for (int j = 0; j < 16; ++j) {
        int k = ((j < 8) ? j : (j + 8)) + (hiK ? 8 : 0);
        b[j] = (m < 2) ? wlds[m * 2304 + (ci0 + k) * 9 + tap] : (_Float16)0.0f;
      }
      acc = __builtin_amdgcn_wmma_f32_16x16x32_f16(false, a, false, b,
                                                   (short)0, acc, false, false);
    }
  }
  // C/D layout: lane<16 -> (M=r, N=lane); lane>=16 -> (M=8+r, N=lane-16)
  if (m < 2) {
    float bv = bias[m];
#pragma unroll
    for (int r = 0; r < 8; ++r) {
      int M = hiK ? (8 + r) : r;
      off[((size_t)bi * 2 + m) * plane + (size_t)y * w2 + (x0 + M)] = tanhf(acc[r] + bv);
    }
  }
}

// ---------------------------------------------------------------- reflect bilinear sample
__global__ __launch_bounds__(256) void grid_sample_kernel(const float* __restrict__ t,
                                                          const float* __restrict__ off,
                                                          float* __restrict__ td,
                                                          int h2, int w2) {
  int plane = h2 * w2;
  int n = BB * C4V * plane;
  int idx = blockIdx.x * blockDim.x + threadIdx.x;
  if (idx >= n) return;
  int x = idx % w2; int tmp = idx / w2;
  int y = tmp % h2; tmp /= h2;
  int ch = tmp % C4V; int bi = tmp / C4V;

  const float* ob = off + (size_t)bi * 2 * plane + (size_t)y * w2 + x;
  float ox = ob[0];
  float oy = ob[(size_t)plane];
  float ix = reflect_clip(((float)x + ox) * ((float)w2 / (float)(w2 - 1)) - 0.5f, w2);
  float iy = reflect_clip(((float)y + oy) * ((float)h2 / (float)(h2 - 1)) - 0.5f, h2);
  float x0f = floorf(ix), y0f = floorf(iy);
  float wx1 = ix - x0f, wy1 = iy - y0f;
  int x0i = (int)x0f, y0i = (int)y0f;
  int x1i = min(x0i + 1, w2 - 1);
  int y1i = min(y0i + 1, h2 - 1);
  const float* tp = t + (size_t)(bi * C4V + ch) * plane;
  // streaming hint: pull the neighborhood of the next row into cache
  __builtin_prefetch(tp + (size_t)min(y0i + 2, h2 - 1) * w2 + x0i, 0, 1);
  float v00 = tp[(size_t)y0i * w2 + x0i];
  float v01 = tp[(size_t)y0i * w2 + x1i];
  float v10 = tp[(size_t)y1i * w2 + x0i];
  float v11 = tp[(size_t)y1i * w2 + x1i];
  float v = v00 * (1.0f - wy1) * (1.0f - wx1) + v01 * (1.0f - wy1) * wx1 +
            v10 * wy1 * (1.0f - wx1) + v11 * wy1 * wx1;
  td[idx] = v;
}

// ---------------------------------------------------------------- depthwise 3x3 + ReLU
__global__ __launch_bounds__(256) void dw3x3_relu(const float* __restrict__ src,
                                                  const float* __restrict__ w,
                                                  float* __restrict__ dst,
                                                  int h2, int w2) {
  int plane = h2 * w2;
  int n = BB * C4V * plane;
  int idx = blockIdx.x * blockDim.x + threadIdx.x;
  if (idx >= n) return;
  int x = idx % w2; int tmp = idx / w2;
  int y = tmp % h2; tmp /= h2;
  int ch = tmp % C4V; int bi = tmp / C4V;
  const float* sp = src + (size_t)(bi * C4V + ch) * plane;
  const float* wp = w + ch * 9;
  // streaming hint: next row below the 3x3 window
  __builtin_prefetch(sp + (size_t)min(y + 2, h2 - 1) * w2 + x, 0, 1);
  float acc = 0.0f;
#pragma unroll
  for (int ky = 0; ky < 3; ++ky) {
    int yy = y + ky - 1;
    if (yy < 0 || yy >= h2) continue;
#pragma unroll
    for (int kx = 0; kx < 3; ++kx) {
      int xx = x + kx - 1;
      if (xx < 0 || xx >= w2) continue;
      acc += wp[ky * 3 + kx] * sp[(size_t)yy * w2 + xx];
    }
  }
  dst[idx] = fmaxf(acc, 0.0f);
}

// ---------------------------------------------------------------- grouped 4->4 pointwise * wscale
__global__ __launch_bounds__(256) void pw_group(const float* __restrict__ s,
                                                const float* __restrict__ pw,
                                                const float* __restrict__ wsc,
                                                float* __restrict__ t2,
                                                int h2, int w2) {
  int plane = h2 * w2;
  int n = BB * CC * plane;  // one thread per (b, group, y, x); does 4 outputs
  int idx = blockIdx.x * blockDim.x + threadIdx.x;
  if (idx >= n) return;
  int x = idx % w2; int tmp = idx / w2;
  int y = tmp % h2; tmp /= h2;
  int g = tmp % CC; int bi = tmp / CC;
  size_t base = ((size_t)bi * C4V + (size_t)g * 4) * plane + (size_t)y * w2 + x;
  float in0 = s[base], in1 = s[base + plane], in2 = s[base + 2 * plane], in3 = s[base + 3 * plane];
#pragma unroll
  for (int o = 0; o < 4; ++o) {
    int ch = g * 4 + o;
    const float* wp = pw + ch * 4;
    float v = wp[0] * in0 + wp[1] * in1 + wp[2] * in2 + wp[3] * in3;
    t2[base + (size_t)o * plane] = v * wsc[ch];
  }
}

// ---------------------------------------------------------------- inverse Haar (no residual add)
// t2: (B, 256, h2, w2) band-interleaved -> out: (B, 64, 2h2, 2w2)
__global__ __launch_bounds__(256) void iwt_kernel(const float* __restrict__ t2,
                                                  float* __restrict__ out,
                                                  int h2, int w2) {
  int H = 2 * h2, Wd = 2 * w2;
  int n = BB * CC * H * Wd;
  int idx = blockIdx.x * blockDim.x + threadIdx.x;
  if (idx >= n) return;
  int X = idx % Wd; int tmp = idx / Wd;
  int Y = tmp % H; tmp /= H;
  int ci = tmp % CC; int bi = tmp / CC;
  int y = Y >> 1, x = X >> 1, ry = Y & 1, rx = X & 1;
  size_t plane = (size_t)h2 * w2;
  size_t tb = ((size_t)bi * C4V + (size_t)ci * 4) * plane + (size_t)y * w2 + x;
  float c0 = t2[tb], c1 = t2[tb + plane], c2 = t2[tb + 2 * plane], c3 = t2[tb + 3 * plane];
  float s1 = ry ? -1.0f : 1.0f;
  float s2 = rx ? -1.0f : 1.0f;
  out[idx] = 0.5f * (c0 + s1 * c1 + s2 * c2 + s1 * s2 * c3);
}

// ---------------------------------------------------------------- fused final:
// out = (dwconv3x3(x)+bias)*scale + gamma * iwt(t2_0 with band0 += nxt1)
__global__ __launch_bounds__(256) void final_kernel(const float* __restrict__ x,
                                                    const float* __restrict__ bw,
                                                    const float* __restrict__ bb,
                                                    const float* __restrict__ bscale,
                                                    const float* __restrict__ t2,
                                                    const float* __restrict__ nxt,
                                                    const float* __restrict__ gamma,
                                                    float* __restrict__ out) {
  const int H = 256, Wd = 256, h2 = 128, w2 = 128;
  int n = BB * CC * H * Wd;
  int idx = blockIdx.x * blockDim.x + threadIdx.x;
  if (idx >= n) return;
  int X = idx % Wd; int tmp = idx / Wd;
  int Y = tmp % H; tmp /= H;
  int ci = tmp % CC; int bi = tmp / CC;

  // depthwise base conv
  const float* sp = x + (size_t)(bi * CC + ci) * H * Wd;
  const float* wp = bw + ci * 9;
  __builtin_prefetch(sp + (size_t)min(Y + 2, H - 1) * Wd + X, 0, 1);
  float acc = 0.0f;
#pragma unroll
  for (int ky = 0; ky < 3; ++ky) {
    int yy = Y + ky - 1;
    if (yy < 0 || yy >= H) continue;
#pragma unroll
    for (int kx = 0; kx < 3; ++kx) {
      int xx = X + kx - 1;
      if (xx < 0 || xx >= Wd) continue;
      acc += wp[ky * 3 + kx] * sp[(size_t)yy * Wd + xx];
    }
  }
  float basev = (acc + bb[ci]) * bscale[ci];

  // inverse Haar with residual on LL band
  int y = Y >> 1, xh = X >> 1, ry = Y & 1, rx = X & 1;
  size_t plane = (size_t)h2 * w2;
  size_t tb = ((size_t)bi * C4V + (size_t)ci * 4) * plane + (size_t)y * w2 + xh;
  float c0 = t2[tb] + nxt[(size_t)(bi * CC + ci) * plane + (size_t)y * w2 + xh];
  float c1 = t2[tb + plane], c2 = t2[tb + 2 * plane], c3 = t2[tb + 3 * plane];
  float s1 = ry ? -1.0f : 1.0f;
  float s2 = rx ? -1.0f : 1.0f;
  float v = 0.5f * (c0 + s1 * c1 + s2 * c2 + s1 * s2 * c3);

  out[idx] = basev + gamma[0] * v;
}

// ---------------------------------------------------------------- launch
extern "C" void kernel_launch(void* const* d_in, const int* in_sizes, int n_in,
                              void* d_out, int out_size, void* d_ws, size_t ws_size,
                              hipStream_t stream) {
  (void)in_sizes; (void)n_in; (void)out_size; (void)ws_size;
  const float* x          = (const float*)d_in[0];
  const float* base_w     = (const float*)d_in[1];
  const float* base_b     = (const float*)d_in[2];
  const float* base_scale = (const float*)d_in[3];
  const float* offset_w   = (const float*)d_in[4];  // (2,2,256,3,3)
  const float* offset_b   = (const float*)d_in[5];  // (2,2)
  const float* dw_w       = (const float*)d_in[6];  // (2,256,1,3,3)
  const float* pw_w       = (const float*)d_in[7];  // (2,256,4,1,1)
  const float* wscale     = (const float*)d_in[8];  // (2,1,256,1,1)
  const float* gamma      = (const float*)d_in[9];  // (1,)
  float* out = (float*)d_out;

  float* ws = (float*)d_ws;
  const size_t S = 16777216;  // 4*256*128*128
  const size_t Q = 4194304;   // 4*256*64*64 == 4*64*128*128
  float* A    = ws;            // t0 -> later s0
  float* Bf   = ws + S;        // td0 -> later {t1, td1, t2_1, nxt1}
  float* Cf   = ws + 2 * S;    // t2_0 (persists to final)
  float* Df   = ws + 3 * S;    // ll1
  float* offb = ws + 3 * S + Q;  // offsets (max 4*2*128*128 = 131072 floats)
  float* t1   = Bf;
  float* td1  = Bf + Q;
  float* t21  = Bf + 2 * Q;
  float* nxt1 = Bf + 3 * Q;

  dim3 blk(256);
  // ---- level 0 (h2=w2=128)
  wt_kernel<<<(BB * CC * 128 * 128 + 255) / 256, blk, 0, stream>>>(x, A, Df, 256, 256);
  offset_conv_wmma<<<(BB * 128 * 128 / 16) / 4, 128, 0, stream>>>(A, offset_w, offset_b, offb, 128, 128);
  grid_sample_kernel<<<(BB * C4V * 128 * 128 + 255) / 256, blk, 0, stream>>>(A, offb, Bf, 128, 128);
  dw3x3_relu<<<(BB * C4V * 128 * 128 + 255) / 256, blk, 0, stream>>>(Bf, dw_w, A, 128, 128);
  pw_group<<<(BB * CC * 128 * 128 + 255) / 256, blk, 0, stream>>>(A, pw_w, wscale, Cf, 128, 128);
  // ---- level 1 (h2=w2=64)
  wt_kernel<<<(BB * CC * 64 * 64 + 255) / 256, blk, 0, stream>>>(Df, t1, nullptr, 128, 128);
  offset_conv_wmma<<<(BB * 64 * 64 / 16) / 4, 128, 0, stream>>>(t1, offset_w + 4608, offset_b + 2, offb, 64, 64);
  grid_sample_kernel<<<(BB * C4V * 64 * 64 + 255) / 256, blk, 0, stream>>>(t1, offb, td1, 64, 64);
  dw3x3_relu<<<(BB * C4V * 64 * 64 + 255) / 256, blk, 0, stream>>>(td1, dw_w + 2304, t1, 64, 64);
  pw_group<<<(BB * CC * 64 * 64 + 255) / 256, blk, 0, stream>>>(t1, pw_w + 1024, wscale + 256, t21, 64, 64);
  // ---- reconstruction
  iwt_kernel<<<(BB * CC * 128 * 128 + 255) / 256, blk, 0, stream>>>(t21, nxt1, 64, 64);
  final_kernel<<<(BB * CC * 256 * 256 + 255) / 256, blk, 0, stream>>>(
      x, base_w, base_b, base_scale, Cf, nxt1, gamma, out);
}
